// CFDSurrogateModel_62440234549301
// MI455X (gfx1250) — matrix operand order repaired and checked by reading, not compiled, not run on hardware
//
#include <hip/hip_runtime.h>
#include <math.h>

#define H 128
#define LAYERS 10

typedef __bf16 bf16_t;
typedef bf16_t v16bf __attribute__((ext_vector_type(16)));
typedef float  v8f   __attribute__((ext_vector_type(8)));

union FragBF {
  v16bf v;
  unsigned int u[8];
};

__device__ __forceinline__ v8f v8f_zero() {
  v8f z = {0.f, 0.f, 0.f, 0.f, 0.f, 0.f, 0.f, 0.f};
  return z;
}

__device__ __forceinline__ unsigned short f2bf(float f) {
  unsigned u = __float_as_uint(f);
  u += 0x7FFFu + ((u >> 16) & 1u);   // round-to-nearest-even
  return (unsigned short)(u >> 16);
}

__device__ __forceinline__ unsigned pack2bf(float a, float b) {
  return (unsigned)f2bf(a) | ((unsigned)f2bf(b) << 16);
}

__device__ __forceinline__ float gelu_exact(float x) {
  return 0.5f * x * (1.0f + erff(x * 0.70710678118654752440f));
}

// A-matrix 16x32 bf16 fragment from LDS (row-major, `stride` ushorts per row).
// ISA 16-bit A layout: lane<16 row M=lane, dw0-3: K=koff+{0,2,4,6}, dw4-7: K=koff+16+{0,2,4,6}, koff=8*(lane>=16)
__device__ __forceinline__ FragBF load_a_frag(const unsigned short* sA, int stride,
                                              int mtile, int kbase, int lane) {
  FragBF f;
  const unsigned short* p = sA + (mtile * 16 + (lane & 15)) * stride
                               + kbase + ((lane >> 4) & 1) * 8;
#pragma unroll
  for (int i = 0; i < 4; ++i) {
    f.u[i]     = *(const unsigned int*)(p + 2 * i);
    f.u[i + 4] = *(const unsigned int*)(p + 16 + 2 * i);
  }
  return f;
}

// B-matrix 32x16 bf16 fragment from transposed weights Wt[N][K] (bf16 bits).
// ISA B layout: N = lane&15, dw0-3: K=kbase+koff+{0,2,4,6}, dw4-7: K=kbase+koff+16+{0,2,4,6}, koff=8*(lane>=16).
__device__ __forceinline__ FragBF load_b_frag(const unsigned short* Wt, int Kdim,
                                              int ntile, int kbase, int lane) {
  FragBF f;
  const unsigned short* p = Wt + (size_t)(ntile * 16 + (lane & 15)) * Kdim
                               + kbase + ((lane >> 4) & 1) * 8;
#pragma unroll
  for (int i = 0; i < 8; ++i) f.u[i] = *(const unsigned int*)(p + 2 * i + (i >> 2) * 8);
  return f;
}

__device__ __forceinline__ v8f wmma_bf16(const FragBF& a, const FragBF& b, v8f c) {
  return __builtin_amdgcn_wmma_f32_16x16x32_bf16(false, a.v, false, b.v, (short)0, c,
                                                 false, false);
}

// Fused MLP block: 64 rows/block, 8 waves (256 thr).
//  A1[64,K1] (gather+bf16) -> GEMM1 -> +b1,LN,GELU -> A2[64,256] -> GEMM2 -> +b2,LN -> residual
// EDGE: A1 = [h[row], h[col], e]; target = e.  !EDGE: A1 = [h, agg*cinv]; target = h.
template <int K1, bool EDGE>
__global__ __launch_bounds__(256) void mlp_block_kernel(
    const float* hsrc, const float* e, const float* agg, const float* cinv,
    const int* __restrict__ row, const int* __restrict__ col,
    const unsigned short* __restrict__ W1t, const float* __restrict__ b1,
    const float* __restrict__ g1, const float* __restrict__ bt1,
    const unsigned short* __restrict__ W2t, const float* __restrict__ b2,
    const float* __restrict__ g2, const float* __restrict__ bt2,
    float* target, int Mtotal) {
  constexpr int M   = 64;       // rows per block
  constexpr int AS  = K1 + 8;   // ushort stride, A1
  constexpr int DS1 = 264;      // float stride, D1 (width 256)
  constexpr int A2S = 264;      // ushort stride, A2 (width 256), overlaid on sA
  constexpr int DS2 = 136;      // float stride, D2 (width 128), overlaid on sD

  __shared__ unsigned short sA[M * AS];     // 64*392*2 = 50176 B (edge)
  __shared__ float sD[M * DS1];             // 64*264*4 = 67584 B
  __shared__ int sRow[M], sCol[M];
  unsigned short* sA2 = sA;

  const int tid   = threadIdx.x;
  const int wave  = tid >> 5;
  const int lane  = tid & 31;
  const int laneN = lane & 15;
  const int laneH = (lane >> 4) & 1;
  const int m0    = blockIdx.x * M;

  if (EDGE) {
    if (tid < M) {
      sRow[tid] = row[m0 + tid];
      sCol[tid] = col[m0 + tid];
    }
    __syncthreads();
  }

  // ---- stage A1 (packed bf16 pairs; segments are 128 wide so pairs never straddle) ----
  for (int i = tid; i < M * (K1 / 2); i += 256) {
    int r  = i / (K1 / 2);
    int c  = (i - r * (K1 / 2)) * 2;
    float v0 = 0.0f, v1 = 0.0f;
    if (EDGE) {
      const float* src;
      if (c < H)          src = hsrc + (size_t)sRow[r] * H + c;
      else if (c < 2 * H) src = hsrc + (size_t)sCol[r] * H + (c - H);
      else                src = e + (size_t)(m0 + r) * H + (c - 2 * H);
      v0 = src[0]; v1 = src[1];
    } else {
      int nidx = m0 + r;
      if (nidx < Mtotal) {
        if (c < H) {
          const float* src = hsrc + (size_t)nidx * H + c;
          v0 = src[0]; v1 = src[1];
        } else {
          const float* src = agg + (size_t)nidx * H + (c - H);
          float ci = cinv[nidx];
          v0 = src[0] * ci; v1 = src[1] * ci;
        }
      }
    }
    *(unsigned*)(sA + r * AS + c) = pack2bf(v0, v1);
  }
  __syncthreads();

  // ---- GEMM1: [64,K1] x [K1,256]; wave owns ntiles {2w,2w+1}, mtiles 0..3 ----
  {
    v8f acc[4][2];
#pragma unroll
    for (int mt = 0; mt < 4; ++mt) {
      acc[mt][0] = v8f_zero();
      acc[mt][1] = v8f_zero();
    }
    const int nt0 = 2 * wave;
#pragma unroll 2
    for (int kt = 0; kt < K1 / 32; ++kt) {
      int kb = kt * 32;
      FragBF b0  = load_b_frag(W1t, K1, nt0, kb, lane);
      FragBF b1f = load_b_frag(W1t, K1, nt0 + 1, kb, lane);
#pragma unroll
      for (int mt = 0; mt < 4; ++mt) {
        FragBF a = load_a_frag(sA, AS, mt, kb, lane);
        acc[mt][0] = wmma_bf16(a, b0, acc[mt][0]);
        acc[mt][1] = wmma_bf16(a, b1f, acc[mt][1]);
      }
    }
#pragma unroll
    for (int mt = 0; mt < 4; ++mt)
#pragma unroll
      for (int nt = 0; nt < 2; ++nt)
#pragma unroll
        for (int r = 0; r < 8; ++r)
          sD[(mt * 16 + r + 8 * laneH) * DS1 + (nt0 + nt) * 16 + laneN] = acc[mt][nt][r];
  }
  __syncthreads();

  // ---- +b1, LN(256), GELU -> A2 bf16; wave handles 8 rows, lane owns 4 column pairs ----
#pragma unroll 1
  for (int rr = 0; rr < 8; ++rr) {
    int rl = wave * 8 + rr;
    float vals[8], s = 0.f, ss = 0.f;
#pragma unroll
    for (int i = 0; i < 4; ++i) {
      int c = 2 * lane + 64 * i;
      float x0 = sD[rl * DS1 + c] + b1[c];
      float x1 = sD[rl * DS1 + c + 1] + b1[c + 1];
      vals[2 * i] = x0; vals[2 * i + 1] = x1;
      s += x0 + x1; ss += x0 * x0 + x1 * x1;
    }
#pragma unroll
    for (int off = 16; off > 0; off >>= 1) {
      s += __shfl_xor(s, off, 32);
      ss += __shfl_xor(ss, off, 32);
    }
    float mean = s * (1.0f / 256.0f);
    float rstd = rsqrtf(ss * (1.0f / 256.0f) - mean * mean + 1e-5f);
#pragma unroll
    for (int i = 0; i < 4; ++i) {
      int c = 2 * lane + 64 * i;
      float y0 = gelu_exact((vals[2 * i] - mean) * rstd * g1[c] + bt1[c]);
      float y1 = gelu_exact((vals[2 * i + 1] - mean) * rstd * g1[c + 1] + bt1[c + 1]);
      *(unsigned*)(sA2 + rl * A2S + c) = pack2bf(y0, y1);
    }
  }
  __syncthreads();

  // ---- GEMM2: [64,256] x [256,128]; wave owns ntile=wave, mtiles 0..3 ----
  {
    v8f acc[4];
#pragma unroll
    for (int mt = 0; mt < 4; ++mt) acc[mt] = v8f_zero();
#pragma unroll 2
    for (int kt = 0; kt < 8; ++kt) {
      int kb = kt * 32;
      FragBF b0 = load_b_frag(W2t, 256, wave, kb, lane);
#pragma unroll
      for (int mt = 0; mt < 4; ++mt) {
        FragBF a = load_a_frag(sA2, A2S, mt, kb, lane);
        acc[mt] = wmma_bf16(a, b0, acc[mt]);
      }
    }
#pragma unroll
    for (int mt = 0; mt < 4; ++mt)
#pragma unroll
      for (int r = 0; r < 8; ++r)
        sD[(mt * 16 + r + 8 * laneH) * DS2 + wave * 16 + laneN] = acc[mt][r];
  }
  __syncthreads();

  // ---- +b2, LN(128), residual writeback; wave handles 8 rows, lane owns 2 column pairs ----
#pragma unroll 1
  for (int rr = 0; rr < 8; ++rr) {
    int rl = wave * 8 + rr;
    int gi = m0 + rl;
    if (!EDGE && gi >= Mtotal) continue;
    float vals[4], s = 0.f, ss = 0.f;
#pragma unroll
    for (int i = 0; i < 2; ++i) {
      int c = 2 * lane + 64 * i;
      float x0 = sD[rl * DS2 + c] + b2[c];
      float x1 = sD[rl * DS2 + c + 1] + b2[c + 1];
      vals[2 * i] = x0; vals[2 * i + 1] = x1;
      s += x0 + x1; ss += x0 * x0 + x1 * x1;
    }
#pragma unroll
    for (int off = 16; off > 0; off >>= 1) {
      s += __shfl_xor(s, off, 32);
      ss += __shfl_xor(ss, off, 32);
    }
    float mean = s * (1.0f / 128.0f);
    float rstd = rsqrtf(ss * (1.0f / 128.0f) - mean * mean + 1e-5f);
#pragma unroll
    for (int i = 0; i < 2; ++i) {
      int c = 2 * lane + 64 * i;
      float* dst = target + (size_t)gi * H + c;
      dst[0] += (vals[2 * i] - mean) * rstd * g2[c] + bt2[c];
      dst[1] += (vals[2 * i + 1] - mean) * rstd * g2[c + 1] + bt2[c + 1];
    }
  }
}

// h = GELU(LN(x @ enc_W + enc_b)); one wave per node, lane = 4 channels
__global__ __launch_bounds__(256) void encoder_kernel(
    const float* __restrict__ x, const float* __restrict__ W, const float* __restrict__ b,
    const float* __restrict__ g, const float* __restrict__ beta, float* __restrict__ h, int N) {
  int wave = threadIdx.x >> 5, lane = threadIdx.x & 31;
  int node = blockIdx.x * 8 + wave;
  if (node >= N) return;
  float acc[4];
#pragma unroll
  for (int i = 0; i < 4; ++i) {
    int c = lane + 32 * i;
    float a = b[c];
#pragma unroll
    for (int k = 0; k < 7; ++k) a += x[node * 7 + k] * W[k * H + c];
    acc[i] = a;
  }
  float s = 0.f, ss = 0.f;
#pragma unroll
  for (int i = 0; i < 4; ++i) { s += acc[i]; ss += acc[i] * acc[i]; }
#pragma unroll
  for (int off = 16; off > 0; off >>= 1) {
    s += __shfl_xor(s, off, 32);
    ss += __shfl_xor(ss, off, 32);
  }
  float mean = s * (1.f / 128.f);
  float rstd = rsqrtf(ss * (1.f / 128.f) - mean * mean + 1e-5f);
#pragma unroll
  for (int i = 0; i < 4; ++i) {
    int c = lane + 32 * i;
    h[(size_t)node * H + c] = gelu_exact((acc[i] - mean) * rstd * g[c] + beta[c]);
  }
}

// e = edge_attr @ eenc_W + eenc_b; one wave per edge
__global__ __launch_bounds__(256) void edge_encoder_kernel(
    const float* __restrict__ ea, const float* __restrict__ W, const float* __restrict__ b,
    float* __restrict__ e, int E) {
  int wave = threadIdx.x >> 5, lane = threadIdx.x & 31;
  int j = blockIdx.x * 8 + wave;
  if (j >= E) return;
#pragma unroll
  for (int i = 0; i < 4; ++i) {
    int c = lane + 32 * i;
    float a = b[c];
#pragma unroll
    for (int k = 0; k < 8; ++k) a += ea[j * 8 + k] * W[k * H + c];
    e[(size_t)j * H + c] = a;
  }
}

__global__ void count_kernel(const int* __restrict__ col, int* __restrict__ cnt, int E) {
  int j = blockIdx.x * blockDim.x + threadIdx.x;
  if (j < E) atomicAdd(&cnt[col[j]], 1);
}

__global__ void cinv_kernel(const int* __restrict__ cnt, float* __restrict__ cinv, int N) {
  int i = blockIdx.x * blockDim.x + threadIdx.x;
  if (i < N) cinv[i] = 1.0f / fmaxf((float)cnt[i], 1.0f);
}

// agg[col[j]] += e[j]; 32 threads/edge, 4 channels each (global_atomic_add_f32)
__global__ void scatter_add_kernel(const float* __restrict__ e, const int* __restrict__ col,
                                   float* __restrict__ agg, int E) {
  int idx = blockIdx.x * blockDim.x + threadIdx.x;
  if (idx >= E * 32) return;
  int j  = idx >> 5;
  int c0 = (idx & 31) * 4;
  int d  = col[j];
  const float* src = e + (size_t)j * H + c0;
  float* dst = agg + (size_t)d * H + c0;
  atomicAdd(dst + 0, src[0]);
  atomicAdd(dst + 1, src[1]);
  atomicAdd(dst + 2, src[2]);
  atomicAdd(dst + 3, src[3]);
}

// fp32 W[layers][K][Nc] -> bf16 Wt[layers][Nc][K] (transposed for B fragments)
__global__ void wconv_kernel(const float* __restrict__ in, unsigned short* __restrict__ out,
                             int K, int Nc, long total) {
  long idx = (long)blockIdx.x * blockDim.x + threadIdx.x;
  if (idx >= total) return;
  long per = (long)K * Nc;
  int l = (int)(idx / per);
  int rem = (int)(idx - (long)l * per);
  int k = rem / Nc;
  int n = rem - k * Nc;
  out[((long)l * Nc + n) * K + k] = f2bf(in[idx]);
}

// out = GELU(h @ dW1 + db1) @ dW2 + db2; one wave per node
__global__ __launch_bounds__(256) void decoder_kernel(
    const float* __restrict__ h, const float* __restrict__ W1, const float* __restrict__ b1,
    const float* __restrict__ W2, const float* __restrict__ b2, float* __restrict__ out, int N) {
  int wave = threadIdx.x >> 5, lane = threadIdx.x & 31;
  int node = blockIdx.x * 8 + wave;
  if (node >= N) return;
  float t[4];
#pragma unroll
  for (int i = 0; i < 4; ++i) {
    int c = lane + 32 * i;
    float a = b1[c];
    for (int k = 0; k < H; ++k) a += h[(size_t)node * H + k] * W1[k * H + c];
    t[i] = gelu_exact(a);
  }
  float o[4] = {0.f, 0.f, 0.f, 0.f};
#pragma unroll
  for (int i = 0; i < 4; ++i) {
    int c = lane + 32 * i;
#pragma unroll
    for (int j = 0; j < 4; ++j) o[j] += t[i] * W2[c * 4 + j];
  }
#pragma unroll
  for (int j = 0; j < 4; ++j)
#pragma unroll
    for (int off = 16; off > 0; off >>= 1) o[j] += __shfl_xor(o[j], off, 32);
  if (lane == 0) {
#pragma unroll
    for (int j = 0; j < 4; ++j) out[node * 4 + j] = o[j] + b2[j];
  }
}

extern "C" void kernel_launch(void* const* d_in, const int* in_sizes, int n_in,
                              void* d_out, int out_size, void* d_ws, size_t ws_size,
                              hipStream_t stream) {
  (void)in_sizes; (void)n_in; (void)out_size; (void)ws_size;
  const int N = 10000, E = 160000, L = LAYERS;

  const float* x      = (const float*)d_in[0];
  const int*   eidx   = (const int*)d_in[1];
  const float* eattr  = (const float*)d_in[2];
  const float* enc_W  = (const float*)d_in[3];
  const float* enc_b  = (const float*)d_in[4];
  const float* enc_g  = (const float*)d_in[5];
  const float* enc_bt = (const float*)d_in[6];
  const float* eenc_W = (const float*)d_in[7];
  const float* eenc_b = (const float*)d_in[8];
  const float* eW1 = (const float*)d_in[9];
  const float* eb1 = (const float*)d_in[10];
  const float* eg1 = (const float*)d_in[11];
  const float* ebt1 = (const float*)d_in[12];
  const float* eW2 = (const float*)d_in[13];
  const float* eb2 = (const float*)d_in[14];
  const float* eg2 = (const float*)d_in[15];
  const float* ebt2 = (const float*)d_in[16];
  const float* nW1 = (const float*)d_in[17];
  const float* nb1 = (const float*)d_in[18];
  const float* ng1 = (const float*)d_in[19];
  const float* nbt1 = (const float*)d_in[20];
  const float* nW2 = (const float*)d_in[21];
  const float* nb2 = (const float*)d_in[22];
  const float* ng2 = (const float*)d_in[23];
  const float* nbt2 = (const float*)d_in[24];
  const float* dW1 = (const float*)d_in[25];
  const float* db1 = (const float*)d_in[26];
  const float* dW2 = (const float*)d_in[27];
  const float* db2 = (const float*)d_in[28];

  const int* row = eidx;
  const int* col = eidx + E;

  // workspace carving (≈ 97 MB)
  char* p = (char*)d_ws;
  auto carve = [&](size_t bytes) -> char* {
    char* r = p;
    p += (bytes + 255) & ~(size_t)255;
    return r;
  };
  float* h    = (float*)carve((size_t)N * H * 4);
  float* e    = (float*)carve((size_t)E * H * 4);
  float* agg  = (float*)carve((size_t)N * H * 4);
  int*   cnt  = (int*)carve((size_t)N * 4);
  float* cinv = (float*)carve((size_t)N * 4);
  unsigned short* eW1t = (unsigned short*)carve((size_t)L * 384 * 256 * 2);
  unsigned short* eW2t = (unsigned short*)carve((size_t)L * 256 * 128 * 2);
  unsigned short* nW1t = (unsigned short*)carve((size_t)L * 256 * 256 * 2);
  unsigned short* nW2t = (unsigned short*)carve((size_t)L * 256 * 128 * 2);

  // bf16 transposed weights
  { long t = (long)L * 384 * 256; wconv_kernel<<<(int)((t + 255) / 256), 256, 0, stream>>>(eW1, eW1t, 384, 256, t); }
  { long t = (long)L * 256 * 128; wconv_kernel<<<(int)((t + 255) / 256), 256, 0, stream>>>(eW2, eW2t, 256, 128, t); }
  { long t = (long)L * 256 * 256; wconv_kernel<<<(int)((t + 255) / 256), 256, 0, stream>>>(nW1, nW1t, 256, 256, t); }
  { long t = (long)L * 256 * 128; wconv_kernel<<<(int)((t + 255) / 256), 256, 0, stream>>>(nW2, nW2t, 256, 128, t); }

  // in-degree (constant across layers)
  hipMemsetAsync(cnt, 0, (size_t)N * 4, stream);
  count_kernel<<<(E + 255) / 256, 256, 0, stream>>>(col, cnt, E);
  cinv_kernel<<<(N + 255) / 256, 256, 0, stream>>>(cnt, cinv, N);

  // encoders
  encoder_kernel<<<(N + 7) / 8, 256, 0, stream>>>(x, enc_W, enc_b, enc_g, enc_bt, h, N);
  edge_encoder_kernel<<<(E + 7) / 8, 256, 0, stream>>>(eattr, eenc_W, eenc_b, e, E);

  const int egrid = E / 64;          // E = 160000 divisible by 64
  const int ngrid = (N + 63) / 64;
  for (int l = 0; l < L; ++l) {
    mlp_block_kernel<384, true><<<egrid, 256, 0, stream>>>(
        h, e, nullptr, nullptr, row, col,
        eW1t + (size_t)l * 256 * 384, eb1 + l * 256, eg1 + l * 256, ebt1 + l * 256,
        eW2t + (size_t)l * 128 * 256, eb2 + l * 128, eg2 + l * 128, ebt2 + l * 128,
        e, E);

    hipMemsetAsync(agg, 0, (size_t)N * H * 4, stream);
    scatter_add_kernel<<<(E * 32 + 255) / 256, 256, 0, stream>>>(e, col, agg, E);

    mlp_block_kernel<256, false><<<ngrid, 256, 0, stream>>>(
        h, nullptr, agg, cinv, nullptr, nullptr,
        nW1t + (size_t)l * 256 * 256, nb1 + l * 256, ng1 + l * 256, nbt1 + l * 256,
        nW2t + (size_t)l * 128 * 256, nb2 + l * 128, ng2 + l * 128, nbt2 + l * 128,
        h, N);
  }

  decoder_kernel<<<(N + 7) / 8, 256, 0, stream>>>(h, dW1, db1, dW2, db2, (float*)d_out, N);
}
